// FlashALiBiAttention_61237643706934
// MI455X (gfx1250) — matrix operand order repaired and checked by reading, not compiled, not run on hardware
//
#include <hip/hip_runtime.h>
#include <hip/hip_bf16.h>

// ---------------------------------------------------------------------------
// Types for WMMA bf16: A/B = v16bf (16 bf16 = 8 VGPRs), C/D = v8f (8 VGPRs)
// ---------------------------------------------------------------------------
typedef __attribute__((ext_vector_type(16))) __bf16 v16bf;
typedef __attribute__((ext_vector_type(8)))  __bf16 v8bf;
typedef __attribute__((ext_vector_type(4)))  __bf16 v4bf;
typedef __attribute__((ext_vector_type(8)))  float  v8f;
typedef __attribute__((ext_vector_type(4)))  int    v4i;

static __device__ __forceinline__ v16bf ld_frag2(const __bf16* p0, const __bf16* p1) {
    union { v16bf v; v8bf h[2]; } u;
    u.h[0] = *(const v8bf*)p0;
    u.h[1] = *(const v8bf*)p1;
    return u.v;
}

static __device__ __forceinline__ v8f wmma_bf16(v16bf a, v16bf b, v8f c) {
    return __builtin_amdgcn_wmma_f32_16x16x32_bf16(
        /*neg_a=*/false, a, /*neg_b=*/false, b,
        /*c_mod=*/(short)0, c, /*reuse_a=*/false, /*reuse_b=*/false);
}

// ---------------------------------------------------------------------------
// Async global->LDS copy (GLOBAL_LOAD_ASYNC_TO_LDS_B128, ASYNCcnt-tracked).
// clang-22 signature takes generic v4i* pointers (src, dst, imm offset, cpol).
// Guarded: falls back to a VGPR bounce if the builtins are absent.
// ---------------------------------------------------------------------------
#if __has_builtin(__builtin_amdgcn_global_load_async_to_lds_b128) && \
    __has_builtin(__builtin_amdgcn_s_wait_asynccnt)
#define ASYNC_LDS 1
#else
#define ASYNC_LDS 0
#endif

static __device__ __forceinline__ void cp_b128_to_lds(const void* gsrc, void* ldst) {
#if ASYNC_LDS
    __builtin_amdgcn_global_load_async_to_lds_b128((v4i*)gsrc, (v4i*)ldst, 0, 0);
#else
    *(v8bf*)ldst = *(const v8bf*)gsrc;
#endif
}

static __device__ __forceinline__ void cp_lds_wait() {
#if ASYNC_LDS
    __builtin_amdgcn_s_wait_asynccnt(0);
#endif
}

// ---------------------------------------------------------------------------
// Kernel 1: fp32 -> bf16 cast (4 elements / thread)
// ---------------------------------------------------------------------------
__global__ __launch_bounds__(256) void cast_f32_bf16(const float* __restrict__ in,
                                                     __bf16* __restrict__ out, int n) {
    int i = (blockIdx.x * 256 + threadIdx.x) * 4;
    if (i + 3 < n) {
        float4 f = *(const float4*)(in + i);
        v4bf o;
        o[0] = (__bf16)f.x; o[1] = (__bf16)f.y; o[2] = (__bf16)f.z; o[3] = (__bf16)f.w;
        *(v4bf*)(out + i) = o;
    }
}

// ---------------------------------------------------------------------------
// Kernel 2: bf16 WMMA GEMM  Y = A[M,K] * W[K,N] + bias
//   MODE 0: write bf16, [B,H,S,hd] head-major layout (for Q/K/V)
//   MODE 1: write fp32, row-major [M,N] (final output projection)
// Tile: 128x128, K-step 32. 256 threads = 8 waves, each wave 64x32 (4x2 tiles).
// ---------------------------------------------------------------------------
template <int MODE>
__global__ __launch_bounds__(256) void gemm_bf16_wmma(
    const __bf16* __restrict__ A, const __bf16* __restrict__ W,
    const float* __restrict__ bias, void* __restrict__ outp,
    int M, int N, int K)
{
    __shared__ __align__(16) __bf16 Asmem[128][32];   // [m][k] row-major
    __shared__ __align__(16) __bf16 Bsmem[128][32];   // [n][k] (transposed)

    const int tid    = threadIdx.x;
    const int lane   = tid & 31;
    const int wave   = tid >> 5;
    const int lh     = lane >> 4;     // lane half (0/1)
    const int lane16 = lane & 15;
    const int m0 = blockIdx.y * 128;
    const int n0 = blockIdx.x * 128;
    const int wm = (wave & 1) * 64;   // wave row offset (4 m-tiles)
    const int wn = (wave >> 1) * 32;  // wave col offset (2 n-tiles)

    v8f acc[4][2];
#pragma unroll
    for (int mt = 0; mt < 4; ++mt)
#pragma unroll
        for (int nt = 0; nt < 2; ++nt) acc[mt][nt] = (v8f){};

    for (int k0 = 0; k0 < K; k0 += 32) {
        __syncthreads();
        // ---- stage A tile: 128x32 bf16, 512 x b128 chunks (async -> LDS)
#pragma unroll
        for (int i = 0; i < 2; ++i) {
            int c   = tid + i * 256;
            int row = c >> 2;
            int kk  = (c & 3) * 8;
            cp_b128_to_lds(&A[(size_t)(m0 + row) * K + k0 + kk], &Asmem[row][kk]);
        }
        // ---- stage W tile transposed: read [k][n] coalesced, write [n][k]
#pragma unroll
        for (int i = 0; i < 2; ++i) {
            int c  = tid + i * 256;
            int kk = c >> 4;
            int n8 = (c & 15) * 8;
            v8bf val = *(const v8bf*)&W[(size_t)(k0 + kk) * N + n0 + n8];
#pragma unroll
            for (int e = 0; e < 8; ++e) Bsmem[n8 + e][kk] = val[e];
        }
        // prefetch next k-tile while compute runs
        if (k0 + 32 < K) {
            __builtin_prefetch(&A[(size_t)(m0 + (tid >> 1)) * K + k0 + 32], 0, 1);
            __builtin_prefetch(&W[(size_t)(k0 + 32 + (tid >> 3)) * N + n0], 0, 1);
        }
        cp_lds_wait();
        __syncthreads();

        // ---- fragments + 8 WMMAs
        v16bf afrag[4];
#pragma unroll
        for (int mt = 0; mt < 4; ++mt) {
            const __bf16* p = &Asmem[wm + mt * 16 + lane16][lh * 8];
            afrag[mt] = ld_frag2(p, p + 16);
        }
        v16bf bfrag[2];
#pragma unroll
        for (int nt = 0; nt < 2; ++nt) {
            const __bf16* p = &Bsmem[wn + nt * 16 + lane16][lh * 8];
            bfrag[nt] = ld_frag2(p, p + 16);
        }
#pragma unroll
        for (int mt = 0; mt < 4; ++mt)
#pragma unroll
            for (int nt = 0; nt < 2; ++nt)
                acc[mt][nt] = wmma_bf16(afrag[mt], bfrag[nt], acc[mt][nt]);
    }

    // ---- epilogue: C/D layout M = r + 8*lh, N = lane16
#pragma unroll
    for (int mt = 0; mt < 4; ++mt)
#pragma unroll
        for (int nt = 0; nt < 2; ++nt)
#pragma unroll
            for (int r = 0; r < 8; ++r) {
                int grow = m0 + wm + mt * 16 + r + 8 * lh;
                int gcol = n0 + wn + nt * 16 + lane16;
                float v = acc[mt][nt][r] + bias[gcol];
                if (MODE == 0) {
                    // [B,H,S,hd]: row = b*2048+s, col = h*128+e
                    __bf16* out = (__bf16*)outp;
                    int bb = grow >> 11, s = grow & 2047;
                    int hh = gcol >> 7,  e = gcol & 127;
                    out[(((size_t)(bb * 16 + hh) * 2048) + s) * 128 + e] = (__bf16)v;
                } else {
                    float* out = (float*)outp;
                    out[(size_t)grow * N + gcol] = v;
                }
            }
}

// ---------------------------------------------------------------------------
// Kernel 3: flash attention with ALiBi + causal.
// Grid (S/64, H, B). Block 128 = 4 waves; each wave owns 16 q-rows.
// Q,K staged row-major [64][128] (async -> LDS); V staged transposed
// [128][64] so P*V B-fragments are contiguous along the key dimension.
// ---------------------------------------------------------------------------
__global__ __launch_bounds__(128) void flash_alibi_attn(
    const __bf16* __restrict__ Qb, const __bf16* __restrict__ Kb,
    const __bf16* __restrict__ Vb, __bf16* __restrict__ Ob)
{
    __shared__ __align__(16) __bf16 Qs[64][128];      // 16 KB
    __shared__ __align__(16) __bf16 Ks[64][128];      // 16 KB
    __shared__ __align__(16) __bf16 Vts[128][64];     // 16 KB (transposed)
    __shared__ __align__(16) __bf16 Ps[4][16][64];    //  8 KB (per-wave P)

    const int tid    = threadIdx.x;
    const int lane   = tid & 31;
    const int w      = tid >> 5;
    const int lh     = lane >> 4;
    const int lane16 = lane & 15;
    const int qt = blockIdx.x, h = blockIdx.y, b = blockIdx.z;
    const int q0 = qt * 64;
    const size_t bh = ((size_t)b * 16 + h) * 2048 * 128;

    // stage Q (64x128) async; completion covered by first in-loop wait+barrier
    const __bf16* Qg = Qb + bh + (size_t)q0 * 128;
#pragma unroll
    for (int i = 0; i < 8; ++i) {
        int c = tid + i * 128;
        int row = c >> 4, e8 = (c & 15) * 8;
        cp_b128_to_lds(&Qg[(size_t)row * 128 + e8], &Qs[row][e8]);
    }

    const float scale = 0.08838834764831845f;          // 1/sqrt(128)
    const float slope = exp2f(-0.5f * (float)(h + 1)); // 2^(-8*(h+1)/16)

    float m_i[8], l_i[8];
    v8f o[8];
#pragma unroll
    for (int r = 0; r < 8; ++r) { m_i[r] = -__builtin_inff(); l_i[r] = 0.f; }
#pragma unroll
    for (int ht = 0; ht < 8; ++ht) o[ht] = (v8f){};

    for (int j = 0; j <= qt; ++j) {
        __syncthreads();
        const __bf16* Kg = Kb + bh + (size_t)j * 64 * 128;
        const __bf16* Vg = Vb + bh + (size_t)j * 64 * 128;
        // stage K row-major (async -> LDS)
#pragma unroll
        for (int i = 0; i < 8; ++i) {
            int c = tid + i * 128;
            int row = c >> 4, e8 = (c & 15) * 8;
            cp_b128_to_lds(&Kg[(size_t)row * 128 + e8], &Ks[row][e8]);
        }
        // stage V transposed: read [key][e] coalesced, write Vts[e][key]
#pragma unroll
        for (int i = 0; i < 8; ++i) {
            int c = tid + i * 128;
            int key = c >> 4, e8 = (c & 15) * 8;
            v8bf val = *(const v8bf*)&Vg[(size_t)key * 128 + e8];
#pragma unroll
            for (int e = 0; e < 8; ++e) Vts[e8 + e][key] = val[e];
        }
        cp_lds_wait();
        __syncthreads();

        // ---- S = Q * K^T : 16 rows x 64 keys, fp32 acc (4 n-tiles)
        v8f sacc[4];
#pragma unroll
        for (int nt = 0; nt < 4; ++nt) sacc[nt] = (v8f){};
#pragma unroll
        for (int ks = 0; ks < 4; ++ks) {          // hd = 128 = 4 * 32
            const __bf16* pa = &Qs[w * 16 + lane16][ks * 32 + lh * 8];
            v16bf aq = ld_frag2(pa, pa + 16);
            v16bf bk[4];
#pragma unroll
            for (int nt = 0; nt < 4; ++nt) {      // batch DS loads, then WMMAs
                const __bf16* pb = &Ks[nt * 16 + lane16][ks * 32 + lh * 8];
                bk[nt] = ld_frag2(pb, pb + 16);
            }
#pragma unroll
            for (int nt = 0; nt < 4; ++nt)
                sacc[nt] = wmma_bf16(aq, bk[nt], sacc[nt]);
        }

        // ---- scale + ALiBi bias + causal mask
#pragma unroll
        for (int nt = 0; nt < 4; ++nt) {
            int k_idx = j * 64 + nt * 16 + lane16;
#pragma unroll
            for (int r = 0; r < 8; ++r) {
                int q_idx = q0 + w * 16 + r + 8 * lh;
                float s = sacc[nt][r] * scale + slope * (float)(k_idx - q_idx);
                sacc[nt][r] = (k_idx > q_idx) ? -__builtin_inff() : s;
            }
        }

        // ---- online softmax; rows live in 16-lane halves -> shfl_xor 1..8
        float fr[8];
#pragma unroll
        for (int r = 0; r < 8; ++r) {
            float mx = fmaxf(fmaxf(sacc[0][r], sacc[1][r]),
                             fmaxf(sacc[2][r], sacc[3][r]));
#pragma unroll
            for (int m = 1; m <= 8; m <<= 1) mx = fmaxf(mx, __shfl_xor(mx, m, 32));
            float m_new = fmaxf(m_i[r], mx);
            float f = __expf(m_i[r] - m_new);
            float rs = 0.f;
#pragma unroll
            for (int nt = 0; nt < 4; ++nt) {
                float p = __expf(sacc[nt][r] - m_new);
                sacc[nt][r] = p;
                rs += p;
            }
#pragma unroll
            for (int m = 1; m <= 8; m <<= 1) rs += __shfl_xor(rs, m, 32);
            l_i[r] = l_i[r] * f + rs;
            m_i[r] = m_new;
            fr[r]  = f;
            // P -> LDS (bf16), C-layout row = r + 8*lh
#pragma unroll
            for (int nt = 0; nt < 4; ++nt)
                Ps[w][r + 8 * lh][nt * 16 + lane16] = (__bf16)sacc[nt][r];
        }

        // rescale O accumulators by exp(m_old - m_new)
#pragma unroll
        for (int ht = 0; ht < 8; ++ht)
#pragma unroll
            for (int r = 0; r < 8; ++r) o[ht][r] *= fr[r];

        __syncthreads();   // P visible in A-fragment order

        // ---- O += P * V : keys 64 = 2 * 32
#pragma unroll
        for (int ks = 0; ks < 2; ++ks) {
            const __bf16* pa = &Ps[w][lane16][ks * 32 + lh * 8];
            v16bf ap = ld_frag2(pa, pa + 16);
#pragma unroll
            for (int ht = 0; ht < 8; ++ht) {
                const __bf16* pb = &Vts[ht * 16 + lane16][ks * 32 + lh * 8];
                v16bf bv = ld_frag2(pb, pb + 16);
                o[ht] = wmma_bf16(ap, bv, o[ht]);
            }
        }
    }

    // ---- normalize and store [B,S,D] bf16 for the O-projection GEMM
#pragma unroll
    for (int ht = 0; ht < 8; ++ht)
#pragma unroll
        for (int r = 0; r < 8; ++r) {
            int s = q0 + w * 16 + r + 8 * lh;
            int d = h * 128 + ht * 16 + lane16;
            float v = o[ht][r] / l_i[r];
            Ob[((size_t)b * 2048 + s) * 2048 + d] = (__bf16)v;
        }
}

// ---------------------------------------------------------------------------
// Host launcher
// ---------------------------------------------------------------------------
extern "C" void kernel_launch(void* const* d_in, const int* in_sizes, int n_in,
                              void* d_out, int out_size, void* d_ws, size_t ws_size,
                              hipStream_t stream) {
    (void)in_sizes; (void)n_in; (void)out_size; (void)ws_size;
    const float* x  = (const float*)d_in[0];
    const float* wq = (const float*)d_in[1];
    const float* bq = (const float*)d_in[2];
    const float* wk = (const float*)d_in[3];
    const float* bk = (const float*)d_in[4];
    const float* wv = (const float*)d_in[5];
    const float* bv = (const float*)d_in[6];
    const float* wo = (const float*)d_in[7];
    const float* bo = (const float*)d_in[8];

    const int M = 4096, N = 2048, K = 2048;       // M = B*S
    const int NX = M * K;                         // 8,388,608
    const int NW = K * N;                         // 4,194,304

    __bf16* xb  = (__bf16*)d_ws;
    __bf16* wqb = xb  + NX;
    __bf16* wkb = wqb + NW;
    __bf16* wvb = wkb + NW;
    __bf16* wob = wvb + NW;
    __bf16* qb  = wob + NW;
    __bf16* kb  = qb  + NX;
    __bf16* vb  = kb  + NX;
    __bf16* ob  = vb  + NX;   // total 58,720,256 bf16 = 112 MB

    // fp32 -> bf16
    cast_f32_bf16<<<NX / 1024, 256, 0, stream>>>(x,  xb,  NX);
    cast_f32_bf16<<<NW / 1024, 256, 0, stream>>>(wq, wqb, NW);
    cast_f32_bf16<<<NW / 1024, 256, 0, stream>>>(wk, wkb, NW);
    cast_f32_bf16<<<NW / 1024, 256, 0, stream>>>(wv, wvb, NW);
    cast_f32_bf16<<<NW / 1024, 256, 0, stream>>>(wo, wob, NW);

    // Q/K/V projections -> bf16 [B,H,S,hd]
    dim3 ggrid(N / 128, M / 128);
    gemm_bf16_wmma<0><<<ggrid, 256, 0, stream>>>(xb, wqb, bq, qb, M, N, K);
    gemm_bf16_wmma<0><<<ggrid, 256, 0, stream>>>(xb, wkb, bk, kb, M, N, K);
    gemm_bf16_wmma<0><<<ggrid, 256, 0, stream>>>(xb, wvb, bv, vb, M, N, K);

    // fused causal ALiBi attention -> bf16 [B,S,D]
    flash_alibi_attn<<<dim3(2048 / 64, 16, 2), 128, 0, stream>>>(qb, kb, vb, ob);

    // output projection -> fp32 d_out
    gemm_bf16_wmma<1><<<ggrid, 256, 0, stream>>>(ob, wob, bo, d_out, M, N, K);
}
